// rnn_decoder_2293512536838
// MI455X (gfx1250) — compile-verified, matrix operand order checked by reading, not run on hardware
//
#include <hip/hip_runtime.h>

// ---------------------------------------------------------------------------
// RNN decoder (2-layer LSTM + Luong attention), MI455X / gfx1250, wave32.
//
// All GEMMs use V_WMMA_F32_16X16X32_BF16 (f32 accumulate). Weights /
// embeddings / contexts are packed once per call into the CDNA5 per-lane
// fragment layouts. The recurrent LSTM kernel stages its shared A-operands
// (x and h fragments) through LDS with GLOBAL_LOAD_ASYNC_TO_LDS_B128
// (ASYNCcnt), and the one-time Ct GEMM bulk-loads its 32KB fragment panels
// with the Tensor Data Mover (TENSOR_LOAD_TO_LDS, TENSORcnt).
// ---------------------------------------------------------------------------

typedef __bf16 bf16_t;
typedef __attribute__((ext_vector_type(16))) __bf16 v16bf;
typedef __attribute__((ext_vector_type(8)))  float  v8f;
typedef __attribute__((ext_vector_type(4)))  unsigned v4u;
typedef __attribute__((ext_vector_type(8)))  int    v8i;
typedef __attribute__((ext_vector_type(4)))  int    v4i;

#define H_   1024
#define B_   32
#define S_   64
#define T_   64
#define L_   2
#define GH_  4096          // 4*H
#define KBH  32            // K-blocks of 32 over K=1024
#define FRAG 512           // halfs per (tile,kb) fragment block: 32 lanes * 16

static __device__ inline v8f wmma_bf16(v16bf a, v16bf b, v8f c) {
  // D = A(16x32) * B(32x16) + C, f32 accum
  return __builtin_amdgcn_wmma_f32_16x16x32_bf16(false, a, false, b,
                                                 (short)0, c, false, false);
}
static __device__ inline float sigm(float x) { return 1.0f / (1.0f + __expf(-x)); }
static __device__ inline v8f zero8() { v8f z = {0.f,0.f,0.f,0.f,0.f,0.f,0.f,0.f}; return z; }

// Generic->LDS byte offset: LDS aperture lives in addr[63:32]; addr[31:0] is
// the in-LDS byte offset (ISA 10.2 aperture mapping).
static __device__ inline unsigned lds_addr32(const void* p) {
  return (unsigned)(size_t)p;
}

// K index within a 32-wide K-block for (lane, half-slot t), per ISA 16-bit
// A/B fragment layout: lanes 0-15 hold K {0..7,16..23}, lanes 16-31 {8..15,24..31}.
static __device__ inline int kk_of(int lane, int t) {
  return t + 8 * ((t >> 3) + (lane >> 4));
}

// Tensor Data Mover: 1D contiguous global->LDS DMA (data_size=8B units).
static __device__ inline void tdm_load_1d(unsigned lds_off, const void* gsrc,
                                          unsigned nbytes) {
  unsigned long long ga = (unsigned long long)(size_t)gsrc;
  unsigned n8 = nbytes >> 3;  // 8-byte units
  v4u g0;
  g0.x = 1u;                                            // count=1 valid descriptor
  g0.y = lds_off;                                       // lds_addr
  g0.z = (unsigned)ga;                                  // global_addr[31:0]
  g0.w = (unsigned)((ga >> 32) & 0x1FFFFFFu) | 0x80000000u;  // ga[56:32] | type=2
  v8i g1;
  g1[0] = (int)(3u << 16);                  // workgroup_mask=0, data_size=3 (8B)
  g1[1] = (int)((n8 & 0xFFFFu) << 16);      // tensor_dim0[15:0]
  g1[2] = (int)(((n8 >> 16) & 0xFFFFu) | (1u << 16)); // tensor_dim0[31:16] | tensor_dim1=1
  g1[3] = (int)((n8 & 0xFFFFu) << 16);      // tensor_dim1 hi=0 | tile_dim0
  g1[4] = 1;                                // tile_dim1=1, tile_dim2=0
  g1[5] = (int)n8;                          // tensor_dim0_stride lo32
  g1[6] = 0;
  g1[7] = 0;
  v4i z4 = {0, 0, 0, 0};
#if __clang_major__ >= 23
  v8i z8 = {0, 0, 0, 0, 0, 0, 0, 0};
  __builtin_amdgcn_tensor_load_to_lds(g0, g1, z4, z4, z8, 0);
#else
  __builtin_amdgcn_tensor_load_to_lds(g0, g1, z4, z4, 0);
#endif
}

// ---------------------------------------------------------------------------
// Packing kernels (run once per call; deterministic)
// ---------------------------------------------------------------------------

__global__ void pack_frag_rm(const float* __restrict__ src, bf16_t* __restrict__ dst,
                             int Rtiles, int KB, int srcStride) {
  int idx = blockIdx.x * blockDim.x + threadIdx.x;  // one per (rt,kb,lane)
  int total = Rtiles * KB * 32;
  if (idx >= total) return;
  int lane = idx & 31;
  int kb   = (idx >> 5) % KB;
  int rt   = (idx >> 5) / KB;
  const float* s = src + (size_t)(rt * 16 + (lane & 15)) * srcStride + kb * 32;
  bf16_t* d = dst + (size_t)idx * 16;
#pragma unroll
  for (int t = 0; t < 16; ++t) d[t] = (bf16_t)s[kk_of(lane, t)];
}

__global__ void pack_frag_km(const float* __restrict__ src, bf16_t* __restrict__ dst,
                             int Rtiles, int KB, int srcStride) {
  int idx = blockIdx.x * blockDim.x + threadIdx.x;
  int total = Rtiles * KB * 32;
  if (idx >= total) return;
  int lane = idx & 31;
  int kb   = (idx >> 5) % KB;
  int rt   = (idx >> 5) / KB;
  int col  = rt * 16 + (lane & 15);
  bf16_t* d = dst + (size_t)idx * 16;
#pragma unroll
  for (int t = 0; t < 16; ++t)
    d[t] = (bf16_t)src[(size_t)(kb * 32 + kk_of(lane, t)) * srcStride + col];
}

__global__ void embed_pack(const int* __restrict__ tokens, const float* __restrict__ emb,
                           bf16_t* __restrict__ xall) {
  int idx = blockIdx.x * blockDim.x + threadIdx.x;  // (st, mt, kb, lane)
  if (idx >= T_ * 2 * KBH * 32) return;
  int lane = idx & 31;
  int kb   = (idx >> 5) & 31;
  int mt   = (idx >> 10) & 1;
  int st   = idx >> 11;
  int row  = mt * 16 + (lane & 15);
  int tok  = tokens[st * B_ + row];
  const float* s = emb + (size_t)tok * H_ + kb * 32;
  bf16_t* d = xall + (size_t)st * (B_ * H_) + ((size_t)(mt * KBH + kb) * 32 + lane) * 16;
#pragma unroll
  for (int t = 0; t < 16; ++t) d[t] = (bf16_t)s[kk_of(lane, t)];
}

__global__ void sbias_kernel(const float* __restrict__ contexts,
                             const float* __restrict__ b_in, float* __restrict__ sbias) {
  int idx = blockIdx.x * blockDim.x + threadIdx.x;
  if (idx >= B_ * S_) return;
  const float* c = contexts + (size_t)idx * H_;
  float a = 0.f;
  for (int k = 0; k < H_; ++k) a += c[k] * b_in[k];
  sbias[idx] = a;
}

// Ct = contexts @ W_in : M=2048, K=1024, N=1024. One wave per 16x16 tile;
// each WG TDM-loads its two contiguous 32KB fragment panels into LDS.
__global__ __launch_bounds__(32) void ct_gemm(const bf16_t* __restrict__ apack,
                                              const bf16_t* __restrict__ bpack,
                                              float* __restrict__ Ct) {
  __shared__ bf16_t lA[KBH * FRAG];   // 32 KiB
  __shared__ bf16_t lB[KBH * FRAG];   // 32 KiB
  int lane = threadIdx.x & 31;
  int mt = blockIdx.x, nt = blockIdx.y;
  const bf16_t* ap = apack + (size_t)mt * (KBH * FRAG);
  const bf16_t* bp = bpack + (size_t)nt * (KBH * FRAG);
  tdm_load_1d(lds_addr32(lA), ap, KBH * FRAG * 2);
  tdm_load_1d(lds_addr32(lB), bp, KBH * FRAG * 2);
  __builtin_amdgcn_s_wait_tensorcnt(0);
  v8f acc = zero8();
#pragma unroll 4
  for (int kb = 0; kb < KBH; ++kb) {
    v16bf a = *(const v16bf*)&lA[kb * FRAG + lane * 16];
    v16bf b = *(const v16bf*)&lB[kb * FRAG + lane * 16];
    acc = wmma_bf16(a, b, acc);
  }
  int m0 = mt * 16 + 8 * (lane >> 4);
  int n  = nt * 16 + (lane & 15);
#pragma unroll
  for (int v = 0; v < 8; ++v) Ct[(size_t)(m0 + v) * H_ + n] = acc[v];
}

// ---------------------------------------------------------------------------
// Per-step kernels
// ---------------------------------------------------------------------------

// One LSTM layer step. grid=64 (16 hidden cols each), block=128 (wave w =
// gate w: i,f,g,o). x/h A-fragments are shared by all 4 waves, so they are
// staged through LDS with async global->LDS loads, double-buffered 2 K-blocks
// at a time. Weights stream directly from L2. Fused bias + cell nonlinearity;
// writes f32 h/c state and next-step bf16 h fragments.
__global__ __launch_bounds__(128) void lstm_step(
    const bf16_t* __restrict__ xfrag, const bf16_t* __restrict__ hfrag_in,
    bf16_t* __restrict__ hfrag_out,
    const bf16_t* __restrict__ wih, const bf16_t* __restrict__ whh,
    const float* __restrict__ bih, const float* __restrict__ bhh,
    float* __restrict__ hst, float* __restrict__ cst) {
  __shared__ bf16_t ab[2][4][1024];   // [buf][chunk: x0,x1,h0,h1][2 kb * 512] = 16KB
  __shared__ float gl[4][2][8][32];   // [gate][mtile][vgpr][lane] = 8KB
  const int lane = threadIdx.x & 31;
  const int wv = threadIdx.x >> 5;                  // gate index / staging chunk
  const int nt = wv * (H_ / 16) + blockIdx.x;       // tile in 4H

  // this wave's staging source: chunk wv = {x mt0, x mt1, h mt0, h mt1}
  const bf16_t* src = (wv < 2) ? xfrag : hfrag_in;
  const unsigned mtoff = (unsigned)(wv & 1) * (KBH * FRAG * 2);  // bytes

  // stage st covers kb = 2*st, 2*st+1 : 2KB contiguous per chunk.
  // NOTE: inst offset applies to BOTH the LDS and global address (ISA 10.7.4).
  auto issue = [&](int st, int q) {
    unsigned goff = mtoff + (unsigned)st * 2048 + (unsigned)lane * 16;
    unsigned ldst = lds_addr32(&ab[q][wv][0]) + (unsigned)lane * 16;
    asm volatile("global_load_async_to_lds_b128 %0, %1, %2 offset:0"
                 :: "v"(ldst), "v"(goff), "s"(src) : "memory");
    asm volatile("global_load_async_to_lds_b128 %0, %1, %2 offset:512"
                 :: "v"(ldst), "v"(goff), "s"(src) : "memory");
    asm volatile("global_load_async_to_lds_b128 %0, %1, %2 offset:1024"
                 :: "v"(ldst), "v"(goff), "s"(src) : "memory");
    asm volatile("global_load_async_to_lds_b128 %0, %1, %2 offset:1536"
                 :: "v"(ldst), "v"(goff), "s"(src) : "memory");
  };

  v8f acc0 = zero8(), acc1 = zero8();
  const bf16_t* bi = wih + (size_t)nt * KBH * FRAG + lane * 16;
  const bf16_t* bh = whh + (size_t)nt * KBH * FRAG + lane * 16;

  issue(0, 0);
  for (int st = 0; st < KBH / 2; ++st) {
    asm volatile("s_wait_asynccnt 0x0" ::: "memory");   // my chunk landed in LDS
    __syncthreads();                                    // everyone's chunk landed
    if (st + 1 < KBH / 2) issue(st + 1, (st + 1) & 1);
    const int q = st & 1;
#pragma unroll
    for (int j = 0; j < 2; ++j) {
      const int kb = st * 2 + j;
      if (kb + 8 < KBH) {
        __builtin_prefetch(bi + (kb + 8) * FRAG, 0, 1);
        __builtin_prefetch(bh + (kb + 8) * FRAG, 0, 1);
      }
      v16bf a0  = *(const v16bf*)&ab[q][0][j * FRAG + lane * 16];
      v16bf a1  = *(const v16bf*)&ab[q][1][j * FRAG + lane * 16];
      v16bf hf0 = *(const v16bf*)&ab[q][2][j * FRAG + lane * 16];
      v16bf hf1 = *(const v16bf*)&ab[q][3][j * FRAG + lane * 16];
      v16bf w0  = *(const v16bf*)(bi + kb * FRAG);
      v16bf w1  = *(const v16bf*)(bh + kb * FRAG);
      acc0 = wmma_bf16(a0, w0, acc0);
      acc1 = wmma_bf16(a1, w0, acc1);
      acc0 = wmma_bf16(hf0, w1, acc0);
      acc1 = wmma_bf16(hf1, w1, acc1);
    }
  }

  // bias (same for all rows of a column)
  int col = blockIdx.x * 16 + (lane & 15);
  float bsum = bih[wv * H_ + col] + bhh[wv * H_ + col];
#pragma unroll
  for (int v = 0; v < 8; ++v) { acc0[v] += bsum; acc1[v] += bsum; }

#pragma unroll
  for (int v = 0; v < 8; ++v) { gl[wv][0][v][lane] = acc0[v]; gl[wv][1][v][lane] = acc1[v]; }
  __syncthreads();

  // pointwise cell update: 512 elements, 4 per thread
#pragma unroll
  for (int i = 0; i < 4; ++i) {
    int e = threadIdx.x + 128 * i;
    int m = e >> 4, n = e & 15;
    int mt = m >> 4, lm = m & 15;
    int l2 = n + ((lm >> 3) << 4);    // D-layout lane
    int vv = lm & 7;                  // D-layout vgpr
    float iv = gl[0][mt][vv][l2];
    float fv = gl[1][mt][vv][l2];
    float gv = gl[2][mt][vv][l2];
    float ov = gl[3][mt][vv][l2];
    int cj = blockIdx.x * 16 + n;
    float cprev = cst[m * H_ + cj];
    float c1 = sigm(fv) * cprev + sigm(iv) * tanhf(gv);
    float h1 = sigm(ov) * tanhf(c1);
    cst[m * H_ + cj] = c1;
    hst[m * H_ + cj] = h1;
    // scatter into next-step A-fragment layout
    int kb = cj >> 5, kv = cj & 31, hi = kv >> 3;
    int flane = lm + ((hi & 1) << 4);
    int ft = (kv & 7) + ((hi >> 1) << 3);
    hfrag_out[((size_t)(mt * KBH + kb) * 32 + flane) * 16 + ft] = (bf16_t)h1;
  }
}

// Attention: scores = h1 . Ct[b], softmax, ctx = w @ contexts[b]; grid = B.
__global__ __launch_bounds__(256) void attn_step(
    const float* __restrict__ Ct, const float* __restrict__ sbias,
    const float* __restrict__ h1, const float* __restrict__ contexts,
    bf16_t* __restrict__ ctx_pack) {
  int b = blockIdx.x;
  __shared__ float sc[S_];
  int s = threadIdx.x >> 2, q = threadIdx.x & 3;    // 4 threads per score
  const float* ct = Ct + (size_t)(b * S_ + s) * H_ + q * 256;
  const float* hv = h1 + (size_t)b * H_ + q * 256;
  float a = 0.f;
  for (int k = 0; k < 256; k += 4) {
    float4 cv = *(const float4*)(ct + k);
    float4 h4 = *(const float4*)(hv + k);
    a += cv.x * h4.x + cv.y * h4.y + cv.z * h4.z + cv.w * h4.w;
  }
  a += __shfl_xor(a, 1);
  a += __shfl_xor(a, 2);
  if (q == 0) sc[s] = a + sbias[b * S_ + s];
  __syncthreads();
  if (threadIdx.x == 0) {
    float mx = -1e30f;
    for (int i = 0; i < S_; ++i) mx = fmaxf(mx, sc[i]);
    float sm = 0.f;
    for (int i = 0; i < S_; ++i) { float e = __expf(sc[i] - mx); sc[i] = e; sm += e; }
    float inv = 1.0f / sm;
    for (int i = 0; i < S_; ++i) sc[i] *= inv;
  }
  __syncthreads();
  int k0 = threadIdx.x * 4;
  float4 acc = {0.f, 0.f, 0.f, 0.f};
  for (int i = 0; i < S_; ++i) {
    float w = sc[i];
    float4 cv = *(const float4*)(contexts + (size_t)(b * S_ + i) * H_ + k0);
    acc.x += w * cv.x; acc.y += w * cv.y; acc.z += w * cv.z; acc.w += w * cv.w;
  }
  int mt = b >> 4, r = b & 15;
  float vals[4] = {acc.x, acc.y, acc.z, acc.w};
#pragma unroll
  for (int i = 0; i < 4; ++i) {
    int cj = k0 + i;
    int kb = cj >> 5, kv = cj & 31, hi = kv >> 3;
    int flane = r + ((hi & 1) << 4);
    int ft = (kv & 7) + ((hi >> 1) << 3);
    ctx_pack[((size_t)(mt * KBH + kb) * 32 + flane) * 16 + ft] = (bf16_t)vals[i];
  }
}

// out = tanh([ctx, h1] @ W_out^T + b_out). grid=64 (N tiles), block=128,
// 4 waves split combined K=2048 (waves 0,1: ctx; 2,3: h1), LDS reduce.
__global__ __launch_bounds__(128) void out_step(
    const bf16_t* __restrict__ ctx_pack, const bf16_t* __restrict__ h1frag,
    const bf16_t* __restrict__ wout, const float* __restrict__ bout,
    float* __restrict__ outp) {
  const int lane = threadIdx.x & 31;
  const int wv = threadIdx.x >> 5;
  const int nt = blockIdx.x;
  v8f acc0 = zero8(), acc1 = zero8();
  const bf16_t* asrc = (wv < 2) ? ctx_pack : h1frag;
  int kb0 = (wv & 1) * 16;  // local kb range within the A source
  const bf16_t* bb = wout + ((size_t)nt * 64 + wv * 16) * FRAG + lane * 16;
#pragma unroll 4
  for (int i = 0; i < 16; ++i) {
    int kb = kb0 + i;
    v16bf a0 = *(const v16bf*)(asrc + ((size_t)kb) * FRAG + lane * 16);
    v16bf a1 = *(const v16bf*)(asrc + ((size_t)(KBH + kb)) * FRAG + lane * 16);
    v16bf wb = *(const v16bf*)(bb + (size_t)i * FRAG);
    acc0 = wmma_bf16(a0, wb, acc0);
    acc1 = wmma_bf16(a1, wb, acc1);
  }
  __shared__ float part[4][2][8][32];
#pragma unroll
  for (int v = 0; v < 8; ++v) { part[wv][0][v][lane] = acc0[v]; part[wv][1][v][lane] = acc1[v]; }
  __syncthreads();
#pragma unroll
  for (int i = 0; i < 4; ++i) {
    int e = threadIdx.x + 128 * i;
    int m = e >> 4, n = e & 15;
    int mt = m >> 4, lm = m & 15;
    int l2 = n + ((lm >> 3) << 4);
    int vv = lm & 7;
    float sum = part[0][mt][vv][l2] + part[1][mt][vv][l2] +
                part[2][mt][vv][l2] + part[3][mt][vv][l2];
    int cj = nt * 16 + n;
    outp[m * H_ + cj] = tanhf(sum + bout[cj]);
  }
}

// ---------------------------------------------------------------------------
// Host launcher
// ---------------------------------------------------------------------------

#define OFF_WIH   ((size_t)0)                       // L*4H*H*2  = 16 MiB
#define OFF_WHH   (OFF_WIH  + (size_t)16777216)
#define OFF_WOUT  (OFF_WHH  + (size_t)16777216)     // H*2H*2    = 4 MiB
#define OFF_WIN   (OFF_WOUT + (size_t)4194304)      // H*H*2     = 2 MiB
#define OFF_CXP   (OFF_WIN  + (size_t)2097152)      // contexts A-pack = 4 MiB
#define OFF_CT    (OFF_CXP  + (size_t)4194304)      // Ct f32    = 8 MiB
#define OFF_XALL  (OFF_CT   + (size_t)8388608)      // T*B*H*2   = 4 MiB
#define OFF_HPK   (OFF_XALL + (size_t)4194304)      // L*2*B*H*2 = 256 KiB
#define OFF_CTXP  (OFF_HPK  + (size_t)262144)       // B*H*2     = 64 KiB
#define OFF_HST   (OFF_CTXP + (size_t)65536)        // L*B*H*4   = 256 KiB
#define OFF_CST   (OFF_HST  + (size_t)262144)       // L*B*H*4
#define OFF_SBIAS (OFF_CST  + (size_t)262144)       // B*S*4 = 8 KiB

extern "C" void kernel_launch(void* const* d_in, const int* in_sizes, int n_in,
                              void* d_out, int out_size, void* d_ws, size_t ws_size,
                              hipStream_t stream) {
  const int*   tokens   = (const int*)d_in[0];
  const float* h0       = (const float*)d_in[1];
  const float* c0       = (const float*)d_in[2];
  const float* contexts = (const float*)d_in[3];
  const float* emb      = (const float*)d_in[4];
  const float* W_ih     = (const float*)d_in[5];
  const float* W_hh     = (const float*)d_in[6];
  const float* b_ih     = (const float*)d_in[7];
  const float* b_hh     = (const float*)d_in[8];
  const float* W_in     = (const float*)d_in[9];
  const float* b_in     = (const float*)d_in[10];
  const float* W_out    = (const float*)d_in[11];
  const float* b_out    = (const float*)d_in[12];

  char* ws = (char*)d_ws;
  bf16_t* WIH  = (bf16_t*)(ws + OFF_WIH);
  bf16_t* WHH  = (bf16_t*)(ws + OFF_WHH);
  bf16_t* WOUT = (bf16_t*)(ws + OFF_WOUT);
  bf16_t* WIN  = (bf16_t*)(ws + OFF_WIN);
  bf16_t* CXP  = (bf16_t*)(ws + OFF_CXP);
  float*  CT   = (float*)(ws + OFF_CT);
  bf16_t* XALL = (bf16_t*)(ws + OFF_XALL);
  bf16_t* HPK  = (bf16_t*)(ws + OFF_HPK);   // [l][parity] blocks of B*H halfs
  bf16_t* CTXP = (bf16_t*)(ws + OFF_CTXP);
  float*  HST  = (float*)(ws + OFF_HST);
  float*  CST  = (float*)(ws + OFF_CST);
  float*  SB   = (float*)(ws + OFF_SBIAS);

  auto hpk = [&](int l, int p) { return HPK + (size_t)(l * 2 + p) * (B_ * H_); };

  // ---- one-time prep ----
  pack_frag_rm<<<2048, 256, 0, stream>>>(W_ih, WIH, L_ * GH_ / 16, KBH, H_);
  pack_frag_rm<<<2048, 256, 0, stream>>>(W_hh, WHH, L_ * GH_ / 16, KBH, H_);
  pack_frag_rm<<<512, 256, 0, stream>>>(W_out, WOUT, H_ / 16, 64, 2 * H_);
  pack_frag_km<<<256, 256, 0, stream>>>(W_in, WIN, H_ / 16, KBH, H_);
  pack_frag_rm<<<512, 256, 0, stream>>>(contexts, CXP, (B_ * S_) / 16, KBH, H_);
  pack_frag_rm<<<8, 256, 0, stream>>>(h0, hpk(0, 0), 2, KBH, H_);
  pack_frag_rm<<<8, 256, 0, stream>>>(h0 + (size_t)B_ * H_, hpk(1, 0), 2, KBH, H_);
  embed_pack<<<512, 256, 0, stream>>>(tokens, emb, XALL);
  sbias_kernel<<<8, 256, 0, stream>>>(contexts, b_in, SB);
  ct_gemm<<<dim3((B_ * S_) / 16, H_ / 16), 32, 0, stream>>>(CXP, WIN, CT);
  hipMemcpyAsync(HST, h0, (size_t)L_ * B_ * H_ * 4, hipMemcpyDeviceToDevice, stream);
  hipMemcpyAsync(CST, c0, (size_t)L_ * B_ * H_ * 4, hipMemcpyDeviceToDevice, stream);

  // ---- sequential decode ----
  const size_t lw = (size_t)GH_ * H_;  // weight halfs per layer
  for (int t = 0; t < T_; ++t) {
    int pr = t & 1, pw = pr ^ 1;
    const bf16_t* xf = XALL + (size_t)t * (B_ * H_);
    lstm_step<<<64, 128, 0, stream>>>(xf, hpk(0, pr), hpk(0, pw),
                                      WIH, WHH, b_ih, b_hh, HST, CST);
    lstm_step<<<64, 128, 0, stream>>>(hpk(0, pw), hpk(1, pr), hpk(1, pw),
                                      WIH + lw, WHH + lw, b_ih + GH_, b_hh + GH_,
                                      HST + (size_t)B_ * H_, CST + (size_t)B_ * H_);
    attn_step<<<32, 256, 0, stream>>>(CT, SB, HST + (size_t)B_ * H_, contexts, CTXP);
    out_step<<<64, 128, 0, stream>>>(CTXP, hpk(1, pw), WOUT, b_out,
                                     (float*)d_out + (size_t)t * B_ * H_);
  }

  // ---- final hT, cT ----
  float* outf = (float*)d_out;
  hipMemcpyAsync(outf + (size_t)T_ * B_ * H_, HST, (size_t)L_ * B_ * H_ * 4,
                 hipMemcpyDeviceToDevice, stream);
  hipMemcpyAsync(outf + (size_t)T_ * B_ * H_ + (size_t)L_ * B_ * H_, CST,
                 (size_t)L_ * B_ * H_ * 4, hipMemcpyDeviceToDevice, stream);
}
